// Self_Attention_60430189854880
// MI455X (gfx1250) — matrix operand order repaired and checked by reading, not compiled, not run on hardware
//
#include <hip/hip_runtime.h>

typedef __attribute__((ext_vector_type(16))) __bf16 v16bf;
typedef __attribute__((ext_vector_type(8)))  float  v8f;
typedef __attribute__((ext_vector_type(4)))  __bf16 v4bf;
typedef __attribute__((ext_vector_type(4)))  int    v4i;
typedef __attribute__((address_space(1))) v4i* v4i_g;  // global int4*
typedef __attribute__((address_space(3))) v4i* v4i_l;  // LDS int4*

#if defined(__gfx1250__) && __has_builtin(__builtin_amdgcn_global_load_async_to_lds_b128)
#define USE_ASYNC_LDS 1
#endif

union Frag16 { v16bf v; uint4 q[2]; };

// copy 16B global -> LDS; async engine (ASYNCcnt-tracked) when available
__device__ __forceinline__ void cp_b128(const __bf16* g, __bf16* l) {
#if defined(USE_ASYNC_LDS)
  __builtin_amdgcn_global_load_async_to_lds_b128((v4i_g)(void*)(__bf16*)g,
                                                 (v4i_l)(void*)l, 0, 0);
#else
  *(uint4*)l = *(const uint4*)g;
#endif
}

template <int N>
__device__ __forceinline__ void wait_async() {
#if defined(USE_ASYNC_LDS)
#if __has_builtin(__builtin_amdgcn_s_wait_asynccnt)
  __builtin_amdgcn_s_wait_asynccnt(N);
#else
  asm volatile("s_wait_asynccnt %0" ::"i"(N) : "memory");
#endif
#endif
}

// DPP row rotate (16-lane row == reduction domain of a half-wave on wave32).
// VALU-only: co-executes with WMMA instead of hitting the LDS unit like ds_bpermute.
template <int CTRL>
__device__ __forceinline__ float dpp_rot(float x) {
  int xi = __builtin_bit_cast(int, x);
  int r = __builtin_amdgcn_update_dpp(xi, xi, CTRL, 0xF, 0xF, true);
  return __builtin_bit_cast(float, r);
}
__device__ __forceinline__ float row_max16(float v) {
  v = fmaxf(v, dpp_rot<0x128>(v));  // row_ror:8
  v = fmaxf(v, dpp_rot<0x124>(v));  // row_ror:4
  v = fmaxf(v, dpp_rot<0x122>(v));  // row_ror:2
  v = fmaxf(v, dpp_rot<0x121>(v));  // row_ror:1
  return v;
}
__device__ __forceinline__ float row_sum16(float v) {
  v += dpp_rot<0x128>(v);
  v += dpp_rot<0x124>(v);
  v += dpp_rot<0x122>(v);
  v += dpp_rot<0x121>(v);
  return v;
}

// A-fragment (16x32 bf16, M x K): lane l -> row = l&15; halves 0-7 from k = 8*(l>>4),
// halves 8-15 from k = 16 + 8*(l>>4). Source row-major, stride ldr.
__device__ __forceinline__ v16bf load_a_frag(const __bf16* base, int ldr, int lane) {
  const int r = lane & 15, kh = lane >> 4;
  Frag16 f;
  f.q[0] = *(const uint4*)(base + r * ldr + 8 * kh);
  f.q[1] = *(const uint4*)(base + r * ldr + 16 + 8 * kh);
  return f.v;
}

// B-fragment (32x16 bf16, K x N) staged as B^T row-major (row n, col k, stride ldr):
// lane l -> col n = l&15; halves 0-7 from k = 8*(l>>4), halves 8-15 from k = 16 + 8*(l>>4).
__device__ __forceinline__ v16bf load_b_frag(const __bf16* base, int ldr, int lane) {
  const int n = lane & 15, kh = lane >> 4;
  Frag16 f;
  f.q[0] = *(const uint4*)(base + n * ldr + 8 * kh);
  f.q[1] = *(const uint4*)(base + n * ldr + 16 + 8 * kh);
  return f.v;
}

__global__ __launch_bounds__(256) void cvt_f32_bf16(const float* __restrict__ in,
                                                    __bf16* __restrict__ out, int n4) {
  int i = blockIdx.x * 256 + threadIdx.x;
  if (i >= n4) return;
  float4 f = ((const float4*)in)[i];
  v4bf o;
  o.x = (__bf16)f.x; o.y = (__bf16)f.y; o.z = (__bf16)f.z; o.w = (__bf16)f.w;
  ((v4bf*)out)[i] = o;
}

// W (Kdim x Ndim f32) -> Wt (Ndim x Kdim bf16), 64x64 LDS tiles
__global__ __launch_bounds__(256) void transpose_cvt(const float* __restrict__ in,
                                                     __bf16* __restrict__ out,
                                                     int Kdim, int Ndim) {
  __shared__ float tile[64][65];
  const int r0 = blockIdx.y * 64, c0 = blockIdx.x * 64;
  const int tid = threadIdx.x;
#pragma unroll
  for (int i = 0; i < 16; ++i) {
    int idx = tid + i * 256;
    int r = idx >> 6, c = idx & 63;
    tile[r][c] = in[(size_t)(r0 + r) * Ndim + c0 + c];
  }
  __syncthreads();
#pragma unroll
  for (int i = 0; i < 16; ++i) {
    int idx = tid + i * 256;
    int c = idx >> 6, r = idx & 63;
    out[(size_t)(c0 + c) * Kdim + r0 + r] = (__bf16)tile[r][c];
  }
}

// C = A(MxK bf16) @ Bt^T + bias. A and Bt are both row-major with K columns.
// Tiles 128x128x32, 8 waves, wave tile 32x64, double-buffered async LDS staging.
// MODE 1: bf16 -> (b,h,t,d); MODE 2: bf16 -> (b,h,d,t); MODE 3: f32 row-major.
#define BM 128
#define BN 128
#define BKK 32
#define LDT 40
template <int MODE>
__global__ __launch_bounds__(256) void gemm_kernel(
    const __bf16* __restrict__ A, const __bf16* __restrict__ Bt,
    const float* __restrict__ bias, void* __restrict__ Cout,
    int M, int N, int K) {
  __shared__ __align__(16) __bf16 lA[2][BM * LDT];
  __shared__ __align__(16) __bf16 lB[2][BN * LDT];
  const int tid = threadIdx.x, lane = tid & 31, w = tid >> 5;
  const int m0 = blockIdx.y * BM, n0 = blockIdx.x * BN;
  const int wm = (w & 3) * 32, wn = (w >> 2) * 64;

  auto issue = [&](int k0, int buf) {
#pragma unroll
    for (int i = 0; i < 2; ++i) {
      int c = tid + i * 256;
      int r = c >> 2, cc = (c & 3) * 8;
      cp_b128(A + (size_t)(m0 + r) * K + k0 + cc, &lA[buf][r * LDT + cc]);
      cp_b128(Bt + (size_t)(n0 + r) * K + k0 + cc, &lB[buf][r * LDT + cc]);
    }
  };

  v8f acc[2][4];
#pragma unroll
  for (int i = 0; i < 2; ++i)
#pragma unroll
    for (int j = 0; j < 4; ++j) acc[i][j] = 0.f;

  issue(0, 0);
  int buf = 0;
  for (int k0 = 0; k0 < K; k0 += BKK) {
    if (k0 + BKK < K) {
      issue(k0 + BKK, buf ^ 1);  // overlap next-tile fetch with this tile's math
      wait_async<4>();           // previous tile (4 older async ops/thread) landed
    } else {
      wait_async<0>();
    }
    __syncthreads();

    v16bf af[2];
#pragma unroll
    for (int i = 0; i < 2; ++i)
      af[i] = load_a_frag(&lA[buf][(wm + i * 16) * LDT], LDT, lane);
    v16bf bfr[4];
#pragma unroll
    for (int j = 0; j < 4; ++j)
      bfr[j] = load_b_frag(&lB[buf][(wn + j * 16) * LDT], LDT, lane);
#pragma unroll
    for (int i = 0; i < 2; ++i)
#pragma unroll
      for (int j = 0; j < 4; ++j)
        acc[i][j] = __builtin_amdgcn_wmma_f32_16x16x32_bf16(
            false, af[i], false, bfr[j], (short)0, acc[i][j], false, false);

    __syncthreads();  // all reads of lds[buf] done before it is refilled
    buf ^= 1;
  }

  const int kh = lane >> 4, ln = lane & 15;
#pragma unroll
  for (int i = 0; i < 2; ++i)
#pragma unroll
    for (int j = 0; j < 4; ++j)
#pragma unroll
      for (int v = 0; v < 8; ++v) {
        int m = m0 + wm + i * 16 + v + 8 * kh;
        int n = n0 + wn + j * 16 + ln;
        float val = acc[i][j][v] + bias[n];
        if constexpr (MODE == 1) {  // (b,h,t,d)
          int b = m >> 11, t = m & 2047, h = n >> 6, d = n & 63;
          ((__bf16*)Cout)[((((size_t)b * 16 + h) * 2048 + t) << 6) + d] = (__bf16)val;
        } else if constexpr (MODE == 2) {  // (b,h,d,t)
          int b = m >> 11, t = m & 2047, h = n >> 6, d = n & 63;
          ((__bf16*)Cout)[(((size_t)b * 16 + h) * 64 + d) * 2048 + t] = (__bf16)val;
        } else {  // f32 row-major
          ((float*)Cout)[(size_t)m * N + n] = val;
        }
      }
}

// Flash-attention: grid (T/128, H, B), 8 waves, wave owns 16 queries.
__global__ __launch_bounds__(256) void attn_kernel(
    const __bf16* __restrict__ Qm, const __bf16* __restrict__ Km,
    const __bf16* __restrict__ Vt, __bf16* __restrict__ Om) {
  constexpr int T = 2048;
  constexpr int LK = 72;
  __shared__ __align__(16) __bf16 lK[2][64 * LK];
  __shared__ __align__(16) __bf16 lV[2][64 * LK];
  __shared__ __align__(16) __bf16 lP[8 * 16 * LK];
  const int tid = threadIdx.x, lane = tid & 31, w = tid >> 5;
  const int ln = lane & 15, kh = lane >> 4;
  const int b = blockIdx.z, h = blockIdx.y, q0 = blockIdx.x * 128;
  const size_t bh = (size_t)(b * 16 + h);
  const __bf16* Qb = Qm + bh * T * 64;
  const __bf16* Kb = Km + bh * T * 64;
  const __bf16* Vb = Vt + bh * 64 * T;
  const int qrow = q0 + w * 16;
  __bf16* lPw = lP + w * 16 * LK;

  auto issue = [&](int kb, int buf) {
#pragma unroll
    for (int i = 0; i < 2; ++i) {
      int c = tid + i * 256;
      int r = c >> 3, cc = (c & 7) * 8;
      cp_b128(Kb + (size_t)(kb + r) * 64 + cc, &lK[buf][r * LK + cc]);
      cp_b128(Vb + (size_t)r * T + kb + cc, &lV[buf][r * LK + cc]);
    }
  };

  // Q fragments live in registers for the whole pass
  v16bf qf[2];
#pragma unroll
  for (int f = 0; f < 2; ++f) {
    Frag16 fr;
    fr.q[0] = *(const uint4*)(Qb + (size_t)(qrow + ln) * 64 + f * 32 + 8 * kh);
    fr.q[1] = *(const uint4*)(Qb + (size_t)(qrow + ln) * 64 + f * 32 + 16 + 8 * kh);
    qf[f] = fr.v;
  }

  v8f accO[4];
#pragma unroll
  for (int j = 0; j < 4; ++j) accO[j] = 0.f;
  float mrow[8], lrow[8];
#pragma unroll
  for (int v = 0; v < 8; ++v) { mrow[v] = -3.0e38f; lrow[v] = 0.f; }
  const float scale = 0.125f;  // 1/sqrt(64)

  issue(0, 0);
  int buf = 0;
  for (int kb = 0; kb < T; kb += 64) {
    if (kb + 64 < T) {
      issue(kb + 64, buf ^ 1);
      wait_async<4>();
    } else {
      wait_async<0>();
    }
    __syncthreads();

    // S(16x64) = Q . K^T : B-fragment column n = K row (kb + j*16 + n)
    v8f s[4];
#pragma unroll
    for (int j = 0; j < 4; ++j) s[j] = 0.f;
#pragma unroll
    for (int j = 0; j < 4; ++j)
#pragma unroll
      for (int f = 0; f < 2; ++f) {
        v16bf bk = load_b_frag(&lK[buf][j * 16 * LK + f * 32], LK, lane);
        s[j] = __builtin_amdgcn_wmma_f32_16x16x32_bf16(
            false, qf[f], false, bk, (short)0, s[j], false, false);
      }

    // online softmax; row m = v + 8*kh spans the 16 lanes of a half-wave,
    // reductions via DPP row rotates (VALU, co-executes with WMMA)
    float fac[8];
#pragma unroll
    for (int v = 0; v < 8; ++v) {
      float cm = fmaxf(fmaxf(s[0][v], s[1][v]), fmaxf(s[2][v], s[3][v])) * scale;
      cm = row_max16(cm);
      float nm = fmaxf(mrow[v], cm);
      fac[v] = __expf(mrow[v] - nm);
      mrow[v] = nm;
      float rs = 0.f;
#pragma unroll
      for (int j = 0; j < 4; ++j) {
        float p = __expf(s[j][v] * scale - nm);
        s[j][v] = p;
        rs += p;
      }
      rs = row_sum16(rs);
      lrow[v] = lrow[v] * fac[v] + rs;
#pragma unroll
      for (int j = 0; j < 4; ++j) accO[j][v] *= fac[v];
    }

    // restage P (D-layout -> A-layout) through per-wave LDS
#pragma unroll
    for (int j = 0; j < 4; ++j)
#pragma unroll
      for (int v = 0; v < 8; ++v)
        lPw[(v + 8 * kh) * LK + j * 16 + ln] = (__bf16)s[j][v];

    asm volatile("s_wait_dscnt 0" ::: "memory");

    // O += P(16x64) . V : B-fragment column n = Vt row (j2*16+n)
#pragma unroll
    for (int f = 0; f < 2; ++f) {
      v16bf pa = load_a_frag(lPw + f * 32, LK, lane);
#pragma unroll
      for (int j2 = 0; j2 < 4; ++j2) {
        v16bf vb = load_b_frag(&lV[buf][j2 * 16 * LK + f * 32], LK, lane);
        accO[j2] = __builtin_amdgcn_wmma_f32_16x16x32_bf16(
            false, pa, false, vb, (short)0, accO[j2], false, false);
      }
    }
    __syncthreads();  // all reads of lds[buf] done before refill
    buf ^= 1;
  }

  // normalize, store O as (b, t, h*64+d) bf16 for the final projection
#pragma unroll
  for (int v = 0; v < 8; ++v) {
    float inv = 1.0f / lrow[v];
    int t = qrow + v + 8 * kh;
#pragma unroll
    for (int j2 = 0; j2 < 4; ++j2) {
      float val = accO[j2][v] * inv;
      int col = h * 64 + j2 * 16 + ln;
      Om[(size_t)(b * T + t) * 1024 + col] = (__bf16)val;
    }
  }
}

extern "C" void kernel_launch(void* const* d_in, const int* in_sizes, int n_in,
                              void* d_out, int out_size, void* d_ws, size_t ws_size,
                              hipStream_t stream) {
  const float* x  = (const float*)d_in[0];
  const float* Wq = (const float*)d_in[1];
  const float* bq = (const float*)d_in[2];
  const float* Wk = (const float*)d_in[3];
  const float* bk = (const float*)d_in[4];
  const float* Wv = (const float*)d_in[5];
  const float* bv = (const float*)d_in[6];
  const float* Wo = (const float*)d_in[7];
  const float* bo = (const float*)d_in[8];

  const int M = 8192, N = 1024, K = 1024;  // M = 4 batches * 2048 tokens
  char* ws = (char*)d_ws;
  size_t off = 0;
  __bf16* xb  = (__bf16*)(ws + off); off += (size_t)M * K * 2;
  __bf16* Wqt = (__bf16*)(ws + off); off += (size_t)K * N * 2;
  __bf16* Wkt = (__bf16*)(ws + off); off += (size_t)K * N * 2;
  __bf16* Wvt = (__bf16*)(ws + off); off += (size_t)K * N * 2;
  __bf16* Wot = (__bf16*)(ws + off); off += (size_t)K * N * 2;
  __bf16* Qb  = (__bf16*)(ws + off); off += (size_t)M * N * 2;
  __bf16* Kb  = (__bf16*)(ws + off); off += (size_t)M * N * 2;
  __bf16* Vtb = (__bf16*)(ws + off); off += (size_t)M * N * 2;
  __bf16* Ob  = (__bf16*)(ws + off); off += (size_t)M * N * 2;

  cvt_f32_bf16<<<M * K / 4 / 256, 256, 0, stream>>>(x, xb, M * K / 4);
  dim3 tg(N / 64, K / 64);
  transpose_cvt<<<tg, 256, 0, stream>>>(Wq, Wqt, K, N);
  transpose_cvt<<<tg, 256, 0, stream>>>(Wk, Wkt, K, N);
  transpose_cvt<<<tg, 256, 0, stream>>>(Wv, Wvt, K, N);
  transpose_cvt<<<tg, 256, 0, stream>>>(Wo, Wot, K, N);

  dim3 gg(N / BN, M / BM);
  gemm_kernel<1><<<gg, 256, 0, stream>>>(xb, Wqt, bq, Qb, M, N, K);
  gemm_kernel<1><<<gg, 256, 0, stream>>>(xb, Wkt, bk, Kb, M, N, K);
  gemm_kernel<2><<<gg, 256, 0, stream>>>(xb, Wvt, bv, Vtb, M, N, K);

  attn_kernel<<<dim3(16, 16, 4), 256, 0, stream>>>(Qb, Kb, Vtb, Ob);

  gemm_kernel<3><<<gg, 256, 0, stream>>>(Ob, Wot, bo, d_out, M, N, K);
}
